// RNN_91216515432819
// MI455X (gfx1250) — matrix operand order repaired and checked by reading, not compiled, not run on hardware
//
#include <hip/hip_runtime.h>
#include <math.h>

// Elman RNN for MI455X (gfx1250, wave32, WMMA, bf16 inputs / f32 accumulate).
//
//  Setup  : emb -> bf16 (flat); Wi,Wh -> transposed bf16 [N][K]; h0 -> bf16.
//  Phase 1: pre = emb_bf16[X] @ Wi + bi  -> d_out (f32)
//  Phase 2: persistent kernel, device-wide barrier per step:
//           h_t = tanh(pre_t + h_{t-1} @ Wh + bh); writes f32 (d_out, in place)
//           and bf16 (ping-pong in ws, feeds next step's A-fragments).

#define T_DIM 1024
#define B_DIM 128
#define DI    256
#define DL    512

typedef __attribute__((ext_vector_type(16))) __bf16 v16bf;
typedef __attribute__((ext_vector_type(8)))  __bf16 v8bf;
typedef __attribute__((ext_vector_type(8)))  float  v8f;

// Workspace layout (bytes)
#define WS_EMBB   0u                      // [V*DI]  bf16 = 5,120,000
#define WS_WIT    5120000u                // [DL][DI] bf16 = 262,144
#define WS_WHT    5382144u                // [DL][DL] bf16 = 524,288
#define WS_HB0    5906432u                // [B][DL]  bf16 = 131,072
#define WS_HB1    6037504u                // [B][DL]  bf16 = 131,072
#define WS_SYNC   6168576u                // 2x u32 barrier counters

// ---- A-fragment: 16-bit A 16x32 (MxK), row-major bf16 source ----
// lanes 0-15 & 16-31 both hold rows M=lane%16; K halves split by lane group:
//   elems 0-7  : K = k0 + (lane<16 ? 0 : 16) .. +7
//   elems 8-15 : same + 8
__device__ __forceinline__ v16bf load_a_bf16(const __bf16* __restrict__ rowp, int lane) {
  const int kk = (lane < 16) ? 0 : 16;
  v8bf lo = *(const v8bf*)(rowp + kk);
  v8bf hi = *(const v8bf*)(rowp + kk + 8);
  return __builtin_shufflevector(lo, hi, 0, 1, 2, 3, 4, 5, 6, 7,
                                         8, 9, 10, 11, 12, 13, 14, 15);
}

// ---- B-fragment: 32x16 (KxN) from [N][K]-transposed bf16 ----
// lane = column (lanes>=16: K+16); 16 contiguous K values per lane.
__device__ __forceinline__ v16bf load_b_t(const __bf16* __restrict__ basT,
                                          int ldk, int col, int k0, int lane) {
  const int koff = (lane < 16) ? 0 : 16;
  return *(const v16bf*)(basT + (size_t)col * ldk + k0 + koff);
}

// =========================  Setup kernels  =========================
__global__ __launch_bounds__(256) void cvt_flat_kernel(
    const float* __restrict__ src, __bf16* __restrict__ dst, int n) {
  const int i = blockIdx.x * 256 + threadIdx.x;
  if (i < n) dst[i] = (__bf16)src[i];
}

// src [K][N] f32 (row-major) -> dst [N][K] bf16
__global__ __launch_bounds__(256) void transpose_cvt_kernel(
    const float* __restrict__ src, __bf16* __restrict__ dst, int K, int N) {
  const int i = blockIdx.x * 256 + threadIdx.x;
  if (i < K * N) {
    const int n = i / K, k = i - n * K;
    dst[i] = (__bf16)src[(size_t)k * N + n];
  }
}

// =====================  Phase 1: pre = emb[X] @ Wi + bi  =====================
// 8192 blocks x 256 threads (8 waves). Wave w: rows [mblock*128+w*16, +16),
// 4 N-tiles (64 cols); A fragment reused 4x.
__global__ __launch_bounds__(256) void rnn_pre_kernel(
    const int*    __restrict__ X,
    const __bf16* __restrict__ embB,   // [V][DI] bf16
    const __bf16* __restrict__ WiT,    // [DL][DI] bf16 (transposed)
    const float*  __restrict__ bi,
    float*        __restrict__ out) {
  const int lane   = threadIdx.x & 31;
  const int wave   = threadIdx.x >> 5;
  const int mblock = blockIdx.x >> 3;
  const int ngroup = blockIdx.x & 7;
  const int row0   = mblock * 128 + wave * 16;
  const int n0     = ngroup * 64;
  const int col    = lane & 15;

  const int tok = X[row0 + (lane & 15)];
  const __bf16* arow = embB + (size_t)tok * DI;

  v8f acc[4] = {v8f{}, v8f{}, v8f{}, v8f{}};

#pragma unroll
  for (int k0 = 0; k0 < DI; k0 += 32) {
    v16bf a = load_a_bf16(arow + k0, lane);
#pragma unroll
    for (int j = 0; j < 4; ++j) {
      v16bf b = load_b_t(WiT, DI, n0 + j * 16 + col, k0, lane);
      acc[j] = __builtin_amdgcn_wmma_f32_16x16x32_bf16(
          false, a, false, b, (short)0, acc[j], false, false);
    }
  }

  const int rbase = (lane < 16) ? 0 : 8;
#pragma unroll
  for (int j = 0; j < 4; ++j) {
    const int n = n0 + j * 16 + col;
    const float bias = bi[n];
#pragma unroll
    for (int r = 0; r < 8; ++r)
      out[(size_t)(row0 + rbase + r) * DL + n] = acc[j][r] + bias;
  }
}

// ==============  Phase 2: persistent sequential scan over T  ==============
// 16 blocks x 512 threads (16 waves). Block owns 32 columns; its 32 KB bf16
// Wh slice (contiguous rows of WhT) is memcpy'd into LDS once. h kept as a
// bf16 ping-pong in ws so A-fragments are pure b128 loads, no conversions.
__global__ __launch_bounds__(512) void rnn_scan_kernel(
    const __bf16* __restrict__ WhT,    // [DL][DL] bf16 (transposed)
    const float*  __restrict__ bh,
    float*        __restrict__ H,      // holds pre on entry; h_t on exit
    __bf16*       __restrict__ hb0,    // ping-pong bf16 h buffers
    __bf16*       __restrict__ hb1,
    unsigned*     __restrict__ sync) {
  __shared__ alignas(32) __bf16 whsT[32 * DL];  // [n_local][k]
  __shared__ float bhs[32];

  const int tid = threadIdx.x;
  const int nb0 = blockIdx.x * 32;

  {  // contiguous copy: rows nb0..nb0+31 of WhT
    const v8bf* src = (const v8bf*)(WhT + (size_t)nb0 * DL);
    v8bf* dst = (v8bf*)whsT;
#pragma unroll
    for (int i = tid; i < (32 * DL) / 8; i += 512) dst[i] = src[i];
  }
  if (tid < 32) bhs[tid] = bh[nb0 + tid];
  __syncthreads();

  const int lane  = tid & 31;
  const int wave  = tid >> 5;
  const int row0  = (wave >> 1) * 16;
  const int ntl   = (wave & 1) * 16;
  const int col   = lane & 15;
  const int rbase = (lane < 16) ? 0 : 8;
  const int ncol  = nb0 + ntl + col;
  const int arow  = row0 + (lane & 15);

  unsigned* count = sync;
  unsigned* gen   = sync + 1;
  const unsigned NB = 16u;

  for (int t = 0; t < T_DIM; ++t) {
    const __bf16* hprev = (t & 1) ? hb1 : hb0;
    __bf16*       hcur  = (t & 1) ? hb0 : hb1;
    float*        cur   = H + (size_t)t * B_DIM * DL;
    const __bf16* ap    = hprev + (size_t)arow * DL;

    v8f acc = {};
    for (int k0 = 0; k0 < DL; k0 += 32) {
      v16bf a = load_a_bf16(ap + k0, lane);
      v16bf b = load_b_t(whsT, DL, ntl + col, k0, lane);
      acc = __builtin_amdgcn_wmma_f32_16x16x32_bf16(
          false, a, false, b, (short)0, acc, false, false);
    }

    const float bias = bhs[ntl + col];
#pragma unroll
    for (int r = 0; r < 8; ++r) {
      const size_t idx = (size_t)(row0 + rbase + r) * DL + ncol;
      const float v = tanhf(acc[r] + cur[idx] + bias);  // pre_t read in place
      cur[idx]  = v;          // f32 output
      hcur[idx] = (__bf16)v;  // bf16 feed for step t+1
    }

    // ---- device-wide barrier: all h_t visible before any block starts t+1 ----
    __syncthreads();
    if (tid == 0) {
      __threadfence();
      const unsigned g = __hip_atomic_load(gen, __ATOMIC_ACQUIRE, __HIP_MEMORY_SCOPE_AGENT);
      const unsigned ticket =
          __hip_atomic_fetch_add(count, 1u, __ATOMIC_ACQ_REL, __HIP_MEMORY_SCOPE_AGENT);
      if (ticket == NB - 1u) {
        __hip_atomic_store(count, 0u, __ATOMIC_RELAXED, __HIP_MEMORY_SCOPE_AGENT);
        __hip_atomic_fetch_add(gen, 1u, __ATOMIC_RELEASE, __HIP_MEMORY_SCOPE_AGENT);
      } else {
        while (__hip_atomic_load(gen, __ATOMIC_ACQUIRE, __HIP_MEMORY_SCOPE_AGENT) == g)
          __builtin_amdgcn_s_sleep(2);
      }
      __threadfence();
    }
    __syncthreads();
  }
}

extern "C" void kernel_launch(void* const* d_in, const int* in_sizes, int n_in,
                              void* d_out, int out_size, void* d_ws, size_t ws_size,
                              hipStream_t stream) {
  (void)in_sizes; (void)n_in; (void)out_size; (void)ws_size;
  const int*   X   = (const int*)  d_in[0];   // [T,B]
  const float* h0  = (const float*)d_in[1];   // [B,DL]
  const float* emb = (const float*)d_in[2];   // [V,DI]
  const float* Wi  = (const float*)d_in[3];   // [DI,DL]
  const float* bi  = (const float*)d_in[4];   // [DL]
  const float* Wh  = (const float*)d_in[5];   // [DL,DL]
  const float* bh  = (const float*)d_in[6];   // [DL]
  float* out = (float*)d_out;                 // [T,B,DL]

  char* ws = (char*)d_ws;
  __bf16* embB = (__bf16*)(ws + WS_EMBB);
  __bf16* WiT  = (__bf16*)(ws + WS_WIT);
  __bf16* WhT  = (__bf16*)(ws + WS_WHT);
  __bf16* hb0  = (__bf16*)(ws + WS_HB0);
  __bf16* hb1  = (__bf16*)(ws + WS_HB1);
  unsigned* sync = (unsigned*)(ws + WS_SYNC);

  hipMemsetAsync(sync, 0, 16, stream);

  // Setup: bf16 conversions / transposes (all reads from d_in only).
  const int n_emb = 10000 * DI;
  cvt_flat_kernel<<<dim3((n_emb + 255) / 256), dim3(256), 0, stream>>>(emb, embB, n_emb);
  cvt_flat_kernel<<<dim3((B_DIM * DL + 255) / 256), dim3(256), 0, stream>>>(h0, hb0, B_DIM * DL);
  transpose_cvt_kernel<<<dim3((DI * DL + 255) / 256), dim3(256), 0, stream>>>(Wi, WiT, DI, DL);
  transpose_cvt_kernel<<<dim3((DL * DL + 255) / 256), dim3(256), 0, stream>>>(Wh, WhT, DL, DL);

  // Phase 1: (T*B/128)*(DL/64) = 8192 blocks.
  rnn_pre_kernel<<<dim3(8192), dim3(256), 0, stream>>>(X, embB, WiT, bi, out);

  // Phase 2: persistent scan, 16 blocks x 16 waves.
  rnn_scan_kernel<<<dim3(16), dim3(512), 0, stream>>>(WhT, bh, out, hb0, hb1, sync);
}
